// PointRendSemSegHead_6347961663931
// MI455X (gfx1250) — matrix operand order repaired and checked by reading, not compiled, not run on hardware
//
#include <hip/hip_runtime.h>
#include <hip/hip_bf16.h>
#include <stdint.h>

typedef __bf16 bf16_t;
typedef __attribute__((ext_vector_type(16))) __bf16 v16bf;
typedef __attribute__((ext_vector_type(8)))  __bf16 v8bf;
typedef __attribute__((ext_vector_type(8)))  float   v8f;

#define NIMG  2
#define NCLS  19
#define CFEAT 256
#define FC    256
#define CIN   275      // CFEAT + NCLS
#define KPAD  288      // CIN padded to multiple of 32 (9 k-steps)
#define NPTS  8192
#define NBINS 1024
#define FH    256
#define FW    512
#define CH    64
#define CW    128

// ---------------------------------------------------------------- helpers

__device__ __forceinline__ int unc_bin(float u) {
    // u = second - first  (<= 0); more uncertain == closer to 0 == smaller bin
    float s = -u * 64.0f;
    int b = (int)s;
    if (b < 0) b = 0;
    if (b > NBINS - 1) b = NBINS - 1;
    return b;
}

// grid_sample(align_corners=False, padding_mode='zeros') on one channel plane
__device__ __forceinline__ float gridsample(const float* __restrict__ ch, int H, int W,
                                            float xc, float yc) {
    float x = xc * (float)W - 0.5f;
    float y = yc * (float)H - 0.5f;
    float x0f = floorf(x), y0f = floorf(y);
    int   x0 = (int)x0f,   y0 = (int)y0f;
    float wx = x - x0f,    wy = y - y0f;
    float acc = 0.0f;
#pragma unroll
    for (int dy = 0; dy < 2; ++dy) {
#pragma unroll
        for (int dx = 0; dx < 2; ++dx) {
            int xi = x0 + dx, yi = y0 + dy;
            float w = (dx ? wx : 1.0f - wx) * (dy ? wy : 1.0f - wy);
            if (xi >= 0 && xi < W && yi >= 0 && yi < H)
                acc += w * ch[yi * W + xi];
        }
    }
    return acc;
}

// async memory->LDS copy of 16 bytes (CDNA5 GLOBAL_LOAD_ASYNC_TO_LDS_B128).
// lds generic pointer: low 32 bits are the LDS byte address (ISA 10.2 aperture).
__device__ __forceinline__ void async_copy_b128(const bf16_t* gsrc, bf16_t* lds_dst) {
    uint32_t loff = (uint32_t)(uintptr_t)lds_dst;
    uint64_t ga   = (uint64_t)(uintptr_t)gsrc;
    asm volatile("global_load_async_to_lds_b128 %0, %1, off"
                 :: "v"(loff), "v"(ga) : "memory");
}

// ---------------------------------------------------------------- kernels

__global__ void zero_i32_kernel(int* __restrict__ p, int count) {
    int i = blockIdx.x * blockDim.x + threadIdx.x;
    if (i < count) p[i] = 0;
}

// fp32 -> bf16 weight conversion with row/K padding
__global__ void convw_kernel(const float* __restrict__ src, bf16_t* __restrict__ dst,
                             int rows_src, int k_src, int rows_dst) {
    int i = blockIdx.x * blockDim.x + threadIdx.x;
    int total = rows_dst * KPAD;
    if (i >= total) return;
    int r = i / KPAD, k = i % KPAD;
    float v = (r < rows_src && k < k_src) ? src[r * k_src + k] : 0.0f;
    dst[i] = (bf16_t)v;
}

// fused: 2x bilinear upsample (half-pixel centers, edge clamp) + top-2 uncertainty
__global__ void upsample_unc_kernel(const float* __restrict__ src, float* __restrict__ dst,
                                    float* __restrict__ unc, int Hs, int Ws, int Hd, int Wd) {
    int i = blockIdx.x * blockDim.x + threadIdx.x;
    int HWd = Hd * Wd;
    if (i >= NIMG * HWd) return;
    int n = i / HWd, pix = i % HWd;
    int oy = pix / Wd, ox = pix % Wd;
    float fy = (oy + 0.5f) * 0.5f - 0.5f;
    float fx = (ox + 0.5f) * 0.5f - 0.5f;
    float y0f = floorf(fy), x0f = floorf(fx);
    float wy = fy - y0f, wx = fx - x0f;
    int y0 = (int)y0f, x0 = (int)x0f;
    int y1 = y0 + 1, x1 = x0 + 1;
    y0 = min(max(y0, 0), Hs - 1); y1 = min(max(y1, 0), Hs - 1);
    x0 = min(max(x0, 0), Ws - 1); x1 = min(max(x1, 0), Ws - 1);
    const float* s = src + (size_t)n * NCLS * Hs * Ws;
    float m1 = -1e30f, m2 = -1e30f;
#pragma unroll 1
    for (int c = 0; c < NCLS; ++c) {
        const float* sc = s + (size_t)c * Hs * Ws;
        float v00 = sc[y0 * Ws + x0], v01 = sc[y0 * Ws + x1];
        float v10 = sc[y1 * Ws + x0], v11 = sc[y1 * Ws + x1];
        float v = (1.0f - wy) * ((1.0f - wx) * v00 + wx * v01)
                +          wy * ((1.0f - wx) * v10 + wx * v11);
        dst[(size_t)n * NCLS * HWd + (size_t)c * HWd + pix] = v;
        if (v > m1) { m2 = m1; m1 = v; } else if (v > m2) { m2 = v; }
    }
    unc[(size_t)n * HWd + pix] = m2 - m1;
}

__global__ void hist_kernel(const float* __restrict__ unc, int* __restrict__ hist, int HW) {
    int i = blockIdx.x * blockDim.x + threadIdx.x;
    if (i >= NIMG * HW) return;
    int n = i / HW;
    atomicAdd(&hist[n * NBINS + unc_bin(unc[i])], 1);
}

__global__ void thresh_kernel(const int* __restrict__ hist, int* __restrict__ thresh) {
    int n = threadIdx.x;
    if (n >= NIMG) return;
    int cum = 0, T = NBINS - 1, before = 0;
    for (int b = 0; b < NBINS; ++b) {
        int h = hist[n * NBINS + b];
        if (cum + h >= NPTS) { T = b; before = cum; break; }
        cum += h;
        before = cum;
    }
    thresh[n * 2 + 0] = T;
    thresh[n * 2 + 1] = before;
}

__global__ void collect_kernel(const float* __restrict__ unc, const int* __restrict__ thresh,
                               int* __restrict__ counters, int* __restrict__ indices, int HW) {
    int i = blockIdx.x * blockDim.x + threadIdx.x;
    if (i >= NIMG * HW) return;
    int n = i / HW, pix = i % HW;
    int b = unc_bin(unc[i]);
    int T = thresh[n * 2], before = thresh[n * 2 + 1];
    if (b < T) {
        int pos = atomicAdd(&counters[n * 2 + 0], 1);
        if (pos < NPTS) indices[n * NPTS + pos] = pix;
    } else if (b == T) {
        int s = atomicAdd(&counters[n * 2 + 1], 1);
        int pos = before + s;
        if (pos < NPTS) indices[n * NPTS + pos] = pix;
    }
}

// Gather fine features (256ch) + coarse logits (19ch) at each point; emit bf16
// activation matrix: point-major rows of KPAD=288 (rows 275..287 zero padded).
// Coarse tail + padding written to BOTH Xa and Xb (concat happens every layer).
__global__ void sample_kernel(const float* __restrict__ feat, const float* __restrict__ coarse,
                              const int* __restrict__ indices,
                              bf16_t* __restrict__ Xa, bf16_t* __restrict__ Xb,
                              int H, int W) {
    int n = blockIdx.x / NPTS;
    int p = blockIdx.x % NPTS;
    int c = threadIdx.x;             // 0..287
    int idx = indices[n * NPTS + p];
    float xc = ((float)(idx % W) + 0.5f) / (float)W;
    float yc = ((float)(idx / W) + 0.5f) / (float)H;
    size_t row = (size_t)(n * NPTS + p) * KPAD;
    if (c < CFEAT) {
        const float* ch = feat + ((size_t)n * CFEAT + c) * (FH * FW);
        Xa[row + c] = (bf16_t)gridsample(ch, FH, FW, xc, yc);
    } else if (c < CIN) {
        int cc = c - CFEAT;
        const float* ch = coarse + ((size_t)n * NCLS + cc) * (CH * CW);
        bf16_t v = (bf16_t)gridsample(ch, CH, CW, xc, yc);
        Xa[row + c] = v;
        Xb[row + c] = v;
    } else if (c < KPAD) {
        Xa[row + c] = (bf16_t)0.0f;
        Xb[row + c] = (bf16_t)0.0f;
    }
}

__device__ __forceinline__ v16bf make_a_frag(const bf16_t* __restrict__ abase) {
    v8bf lo = *(const v8bf*)(abase);
    v8bf hi = *(const v8bf*)(abase + 8);
    v16bf a;
#pragma unroll
    for (int r = 0; r < 8; ++r) { a[r] = lo[r]; a[r + 8] = hi[r]; }
    return a;
}

// Hidden layer: Y[0:256, :] = relu(W(256x288) * X(288xNPTS) + b), bf16 out into
// next activation buffer (rows 256..287 already hold coarse+pad).
// Block: 8 waves; wave = 16 rows x 128 cols (8 WMMA per A-frag).
// Block tile M=128, N=128. LDS B tiles are double-buffered and filled with
// GLOBAL_LOAD_ASYNC_TO_LDS_B128, overlapping the copy of k-step i+1 with the
// WMMAs of k-step i (ASYNCcnt-tracked, async loads complete in order).
#define HBN 128
__global__ void __launch_bounds__(256) gemm_hidden_kernel(
    const bf16_t* __restrict__ Wmat, const bf16_t* __restrict__ X,
    const float* __restrict__ bias, bf16_t* __restrict__ Y) {
    __shared__ bf16_t Bs[2][HBN * 32];
    int tid  = threadIdx.x;
    int wave = tid >> 5, lane = tid & 31;
    int half = lane >> 4, l15 = lane & 15;
    int n  = blockIdx.z;
    int n0 = blockIdx.x * HBN;
    int m0 = blockIdx.y * 128 + wave * 16;
    const bf16_t* Xn = X + (size_t)n * NPTS * KPAD;
    bf16_t*       Yn = Y + (size_t)n * NPTS * KPAD;

    // issue async copies for one 128x32 B tile: 512 chunks of 16B, 2 per thread
    auto issue = [&](int s, int k0) {
#pragma unroll
        for (int i = 0; i < 2; ++i) {
            int t = tid + i * 256;
            int point = t >> 2, chunk = t & 3;
            async_copy_b128(Xn + (size_t)(n0 + point) * KPAD + k0 + chunk * 8,
                            &Bs[s][point * 32 + chunk * 8]);
        }
    };

    issue(0, 0);
    v8f acc[8] = {v8f{}, v8f{}, v8f{}, v8f{}, v8f{}, v8f{}, v8f{}, v8f{}};
    for (int kt = 0; kt < KPAD / 32; ++kt) {
        int k0 = kt * 32;
        if (kt < KPAD / 32 - 1) {
            issue((kt + 1) & 1, k0 + 32);
            asm volatile("s_wait_asynccnt 0x2" ::: "memory");  // stage kt done
        } else {
            asm volatile("s_wait_asynccnt 0x0" ::: "memory");
        }
        __syncthreads();
        const bf16_t* Bb = Bs[kt & 1];
        // A fragment (16x32): lanes 0-15 hold K 0-15; lanes 16-31 K 16-31, contiguous
        v16bf a = make_a_frag(Wmat + (size_t)(m0 + l15) * KPAD + k0 + half * 16);
#pragma unroll
        for (int j = 0; j < 8; ++j) {
            // B fragment (32x16): lanes 0-15 K0-15, lanes 16-31 K16-31, contiguous
            v16bf b = *(const v16bf*)(&Bb[(j * 16 + l15) * 32 + half * 16]);
            acc[j] = __builtin_amdgcn_wmma_f32_16x16x32_bf16(
                false, a, false, b, (short)0, acc[j], false, false);
        }
        __syncthreads();  // buf reused by the async issue in the next iteration
    }
    // epilogue: bias + relu + bf16 store into next activation buffer
#pragma unroll
    for (int j = 0; j < 8; ++j) {
        int p  = n0 + j * 16 + l15;
        int mb = m0 + half * 8;
        v8bf outv;
#pragma unroll
        for (int r = 0; r < 8; ++r) {
            float f = acc[j][r] + bias[mb + r];
            f = f > 0.0f ? f : 0.0f;
            outv[r] = (bf16_t)f;
        }
        *(v8bf*)(Yn + (size_t)p * KPAD + mb) = outv;
    }
}

// Final layer: logits = Wp(32x288, rows>=19 zero) * X + bp, fp32 scatter into
// the upsampled grid through point indices. Block: 8 waves as 2(M) x 4(N);
// block tile M=32, N=256. Same async double-buffered LDS pipeline.
__global__ void __launch_bounds__(256) gemm_final_kernel(
    const bf16_t* __restrict__ Wmat, const bf16_t* __restrict__ X,
    const float* __restrict__ bias, const int* __restrict__ indices,
    float* __restrict__ outbuf, int HW) {
    __shared__ bf16_t Bs[2][256 * 32];
    int tid  = threadIdx.x;
    int wave = tid >> 5, lane = tid & 31;
    int half = lane >> 4, l15 = lane & 15;
    int mi = wave >> 2, nj = wave & 3;
    int n  = blockIdx.z;
    int n0 = blockIdx.x * 256;
    int m0 = mi * 16;
    const bf16_t* Xn = X + (size_t)n * NPTS * KPAD;

    auto issue = [&](int s, int k0) {   // 1024 chunks of 16B, 4 per thread
#pragma unroll
        for (int i = 0; i < 4; ++i) {
            int t = tid + i * 256;
            int point = t >> 2, chunk = t & 3;
            async_copy_b128(Xn + (size_t)(n0 + point) * KPAD + k0 + chunk * 8,
                            &Bs[s][point * 32 + chunk * 8]);
        }
    };

    issue(0, 0);
    v8f acc[4] = {v8f{}, v8f{}, v8f{}, v8f{}};
    for (int kt = 0; kt < KPAD / 32; ++kt) {
        int k0 = kt * 32;
        if (kt < KPAD / 32 - 1) {
            issue((kt + 1) & 1, k0 + 32);
            asm volatile("s_wait_asynccnt 0x4" ::: "memory");
        } else {
            asm volatile("s_wait_asynccnt 0x0" ::: "memory");
        }
        __syncthreads();
        const bf16_t* Bb = Bs[kt & 1];
        v16bf a = make_a_frag(Wmat + (size_t)(m0 + l15) * KPAD + k0 + half * 16);
#pragma unroll
        for (int j = 0; j < 4; ++j) {
            v16bf b = *(const v16bf*)(&Bb[(nj * 64 + j * 16 + l15) * 32 + half * 16]);
            acc[j] = __builtin_amdgcn_wmma_f32_16x16x32_bf16(
                false, a, false, b, (short)0, acc[j], false, false);
        }
        __syncthreads();
    }
#pragma unroll
    for (int j = 0; j < 4; ++j) {
        int p   = n0 + nj * 64 + j * 16 + l15;
        int idx = indices[n * NPTS + p];
#pragma unroll
        for (int r = 0; r < 8; ++r) {
            int m = m0 + half * 8 + r;
            if (m < NCLS)
                outbuf[(size_t)n * NCLS * HW + (size_t)m * HW + idx] = acc[j][r] + bias[m];
        }
    }
}

// ---------------------------------------------------------------- launch

extern "C" void kernel_launch(void* const* d_in, const int* in_sizes, int n_in,
                              void* d_out, int out_size, void* d_ws, size_t ws_size,
                              hipStream_t stream) {
    const float* coarse = (const float*)d_in[0];   // (2,19,64,128)
    const float* feat   = (const float*)d_in[1];   // (2,256,256,512)
    const float* w1 = (const float*)d_in[2]; const float* b1 = (const float*)d_in[3];
    const float* w2 = (const float*)d_in[4]; const float* b2 = (const float*)d_in[5];
    const float* w3 = (const float*)d_in[6]; const float* b3 = (const float*)d_in[7];
    const float* wp = (const float*)d_in[8]; const float* bp = (const float*)d_in[9];
    float* out = (float*)d_out;                    // (2,19,512,1024)

    char* ws = (char*)d_ws;
    size_t off = 0;
    auto alloc = [&](size_t bytes) -> void* {
        void* p = (void*)(ws + off);
        off += (bytes + 255) & ~(size_t)255;
        return p;
    };
    float*  L1       = (float*)alloc((size_t)NIMG * NCLS * 128 * 256 * 4);
    float*  L2       = (float*)alloc((size_t)NIMG * NCLS * 256 * 512 * 4);
    float*  unc      = (float*)alloc((size_t)NIMG * 512 * 1024 * 4);
    int*    hist     = (int*)alloc((size_t)NIMG * NBINS * 4);
    int*    thresh   = (int*)alloc((size_t)NIMG * 2 * 4);
    int*    counters = (int*)alloc((size_t)NIMG * 2 * 4);
    int*    indices  = (int*)alloc((size_t)NIMG * NPTS * 4);
    bf16_t* Xa       = (bf16_t*)alloc((size_t)NIMG * NPTS * KPAD * 2);
    bf16_t* Xb       = (bf16_t*)alloc((size_t)NIMG * NPTS * KPAD * 2);
    bf16_t* Wb       = (bf16_t*)alloc((size_t)(3 * FC + 32) * KPAD * 2);
    bf16_t* W1b = Wb;
    bf16_t* W2b = Wb + (size_t)1 * FC * KPAD;
    bf16_t* W3b = Wb + (size_t)2 * FC * KPAD;
    bf16_t* Wpb = Wb + (size_t)3 * FC * KPAD;

    // convert weights to bf16, pad K to 288 (and wp rows 19->32)
    {
        int tW = FC * KPAD;
        convw_kernel<<<(tW + 255) / 256, 256, 0, stream>>>(w1, W1b, FC, CIN, FC);
        convw_kernel<<<(tW + 255) / 256, 256, 0, stream>>>(w2, W2b, FC, CIN, FC);
        convw_kernel<<<(tW + 255) / 256, 256, 0, stream>>>(w3, W3b, FC, CIN, FC);
        int tP = 32 * KPAD;
        convw_kernel<<<(tP + 255) / 256, 256, 0, stream>>>(wp, Wpb, NCLS, CIN, 32);
    }

    struct Level { const float* src; float* dst; int Hs, Ws; };
    Level lv[3] = { {coarse, L1, 64, 128}, {L1, L2, 128, 256}, {L2, out, 256, 512} };

    for (int s = 0; s < 3; ++s) {
        int Hd = lv[s].Hs * 2, Wd = lv[s].Ws * 2;
        int HW = Hd * Wd;
        int total = NIMG * HW;

        upsample_unc_kernel<<<(total + 255) / 256, 256, 0, stream>>>(
            lv[s].src, lv[s].dst, unc, lv[s].Hs, lv[s].Ws, Hd, Wd);

        zero_i32_kernel<<<(NIMG * NBINS + 255) / 256, 256, 0, stream>>>(hist, NIMG * NBINS);
        zero_i32_kernel<<<1, 32, 0, stream>>>(counters, NIMG * 2);

        hist_kernel<<<(total + 255) / 256, 256, 0, stream>>>(unc, hist, HW);
        thresh_kernel<<<1, 32, 0, stream>>>(hist, thresh);
        collect_kernel<<<(total + 255) / 256, 256, 0, stream>>>(unc, thresh, counters, indices, HW);

        sample_kernel<<<NIMG * NPTS, KPAD, 0, stream>>>(feat, coarse, indices, Xa, Xb, Hd, Wd);

        gemm_hidden_kernel<<<dim3(NPTS / HBN, FC / 128, NIMG), 256, 0, stream>>>(W1b, Xa, b1, Xb);
        gemm_hidden_kernel<<<dim3(NPTS / HBN, FC / 128, NIMG), 256, 0, stream>>>(W2b, Xb, b2, Xa);
        gemm_hidden_kernel<<<dim3(NPTS / HBN, FC / 128, NIMG), 256, 0, stream>>>(W3b, Xa, b3, Xb);
        gemm_final_kernel<<<dim3(NPTS / 256, 1, NIMG), 256, 0, stream>>>(
            Wpb, Xb, bp, indices, lv[s].dst, HW);
    }
}
